// LSTMVectorEstimationModel_33638183862923
// MI455X (gfx1250) — matrix operand order repaired and checked by reading, not compiled, not run on hardware
//
#include <hip/hip_runtime.h>
#include <hip/hip_bf16.h>
#include <cmath>

// LSTM: B=256, T=33, E=1024, L=32 steps, per-step weights (L,4E,E).
// HBM-bound on streaming 1.07GB of fp32 weights once; compute via
// v_wmma_f32_16x16x32_f16 with in-register fp32->f16 conversion.
// Each block = 8 waves cooperating on one 16x16 h/c tile with an
// 8-way K-split (boosts wave parallelism 8x for latency hiding),
// reduced through LDS.

#define B_ 256
#define T_ 33
#define E_ 1024
#define L_ 32

typedef __attribute__((ext_vector_type(16))) _Float16 v16h;
typedef __attribute__((ext_vector_type(8)))  float    v8f;

// Convert two aligned runs of 8 consecutive fp32 into a v16h
// (elements 0..7 from plo, 8..15 from phi).
__device__ __forceinline__ v16h cvt8x2(const float* __restrict__ plo,
                                       const float* __restrict__ phi) {
  const float4 f0 = *(const float4*)(plo + 0);
  const float4 f1 = *(const float4*)(plo + 4);
  const float4 f2 = *(const float4*)(phi + 0);
  const float4 f3 = *(const float4*)(phi + 4);
  v16h r;
  r[0]  = (_Float16)f0.x; r[1]  = (_Float16)f0.y; r[2]  = (_Float16)f0.z; r[3]  = (_Float16)f0.w;
  r[4]  = (_Float16)f1.x; r[5]  = (_Float16)f1.y; r[6]  = (_Float16)f1.z; r[7]  = (_Float16)f1.w;
  r[8]  = (_Float16)f2.x; r[9]  = (_Float16)f2.y; r[10] = (_Float16)f2.z; r[11] = (_Float16)f2.w;
  r[12] = (_Float16)f3.x; r[13] = (_Float16)f3.y; r[14] = (_Float16)f3.z; r[15] = (_Float16)f3.w;
  return r;
}

__device__ __forceinline__ float sigmoid_(float x) {
  return 1.0f / (1.0f + __expf(-x));
}
__device__ __forceinline__ float tanh_(float x) {
  x = fminf(20.0f, fmaxf(-20.0f, x));
  const float e = __expf(2.0f * x);
  return (e - 1.0f) / (e + 1.0f);
}

// Grid: 1024 blocks = (B/16)*(E/16) tiles; block = 256 threads = 8 waves.
// Wave w accumulates K-slice [128w, 128w+128) of all four gate GEMMs
// (x@W_ih^T and h@W_hh^T), partials reduced via LDS, fused cell epilogue.
__global__ __launch_bounds__(256) void lstm_step_kernel(
    const float* __restrict__ xt,     // x + t*E ; row stride T_*E_
    const float* __restrict__ wih,    // W_ih + t*4*E*E ; row stride E_
    const float* __restrict__ whh,    // W_hh + t*4*E*E
    const float* __restrict__ hprev,  // (B,E)
    const float* __restrict__ cprev,  // (B,E)
    float* __restrict__ hnext,
    float* __restrict__ cnext) {
  __shared__ float red[8 * 4 * 8 * 32];  // [wave][gate][elem][lane] = 32KB

  const int tid  = threadIdx.x;
  const int lane = tid & 31;
  const int n    = lane & 15;   // N column / A row (M)
  const int hi   = lane >> 4;   // lane-half selector
  const int w    = tid >> 5;    // wave index in block -> K slice

  const int bi = blockIdx.x >> 6;   // 16 batch tiles
  const int jj = blockIdx.x & 63;   // 64 column tiles
  const int b0 = bi << 4;
  const int j0 = jj << 4;

  const float* xrow = xt    + (size_t)(b0 + n) * (T_ * E_);
  const float* hrow = hprev + (size_t)(b0 + n) * E_;

  const float* wih_r[4];
  const float* whh_r[4];
#pragma unroll
  for (int g = 0; g < 4; ++g) {
    const size_t roff = (size_t)(g * E_ + j0 + n) * E_;
    wih_r[g] = wih + roff;
    whh_r[g] = whh + roff;
  }

  const v8f zero = {0.f, 0.f, 0.f, 0.f, 0.f, 0.f, 0.f, 0.f};
  v8f acc[4];
#pragma unroll
  for (int g = 0; g < 4; ++g) acc[g] = zero;

  const int k_lo = w * (E_ / 8);        // 128-wide K slice per wave
  const int k_hi = k_lo + (E_ / 8);
#pragma unroll
  for (int k = k_lo; k < k_hi; k += 32) {
    const int ka = k + 16 * hi;  // A (16x32 f16) element layout
    const int kb = k + 16 * hi;  // B (32x16 f16) element layout
    const v16h ax = cvt8x2(xrow + ka, xrow + ka + 8);
    const v16h ah = cvt8x2(hrow + ka, hrow + ka + 8);
#pragma unroll
    for (int g = 0; g < 4; ++g) {
      const v16h bI = cvt8x2(wih_r[g] + kb, wih_r[g] + kb + 8);
      acc[g] = __builtin_amdgcn_wmma_f32_16x16x32_f16(
          false, ax, false, bI, (short)0, acc[g], false, false);
      const v16h bH = cvt8x2(whh_r[g] + kb, whh_r[g] + kb + 8);
      acc[g] = __builtin_amdgcn_wmma_f32_16x16x32_f16(
          false, ah, false, bH, (short)0, acc[g], false, false);
    }
  }

  // Publish partial accumulators: conflict-free (lane-major innermost).
#pragma unroll
  for (int g = 0; g < 4; ++g) {
#pragma unroll
    for (int e = 0; e < 8; ++e) {
      red[((w * 4 + g) * 8 + e) * 32 + lane] = acc[g][e];
    }
  }
  __syncthreads();

  // 256 threads <-> 256 tile elements: tid = (e<<5)|lane maps to
  // (M = e + 8*(lane>>4), N = lane&15) per the WMMA C/D layout.
  const int e_r = tid >> 5;
  float gate[4];
#pragma unroll
  for (int g = 0; g < 4; ++g) {
    float s = 0.0f;
#pragma unroll
    for (int ww = 0; ww < 8; ++ww) {
      s += red[((ww * 4 + g) * 8 + e_r) * 32 + lane];
    }
    gate[g] = s;
  }

  const size_t idx = (size_t)(b0 + e_r + 8 * hi) * E_ + (size_t)(j0 + n);
  const float iv = sigmoid_(gate[0]);
  const float fv = sigmoid_(gate[1]);
  const float gv = tanh_(gate[2]);
  const float ov = sigmoid_(gate[3]);
  const float cn = fv * cprev[idx] + iv * gv;
  cnext[idx] = cn;
  hnext[idx] = ov * tanh_(cn);
}

__global__ void lstm_init_kernel(const float* __restrict__ x,
                                 float* __restrict__ h0,
                                 float* __restrict__ c0) {
  const int i = blockIdx.x * 256 + threadIdx.x;  // i < B*E
  const int b = i >> 10;
  const int e = i & (E_ - 1);
  h0[i] = 0.0f;
  c0[i] = x[(size_t)b * (T_ * E_) + E_ + e];  // x[:, 1, :]
}

// Copy final h,c into d_out and produce per-block partial sums of |h - tgt|.
__global__ void lstm_finalize_kernel(const float* __restrict__ hf,
                                     const float* __restrict__ cf,
                                     const float* __restrict__ tgt,
                                     float* __restrict__ out,
                                     float* __restrict__ partials) {
  __shared__ float sm[256];
  const int i = blockIdx.x * 256 + threadIdx.x;  // i < B*E
  const float h = hf[i];
  out[1 + i] = h;
  out[1 + B_ * E_ + i] = cf[i];
  sm[threadIdx.x] = fabsf(h - tgt[i]);
  __syncthreads();
#pragma unroll
  for (int s = 128; s > 0; s >>= 1) {
    if (threadIdx.x < s) sm[threadIdx.x] += sm[threadIdx.x + s];
    __syncthreads();
  }
  if (threadIdx.x == 0) partials[blockIdx.x] = sm[0];
}

__global__ void lstm_reduce_kernel(const float* __restrict__ partials,
                                   float* __restrict__ out) {
  __shared__ float sm[256];
  float s = 0.0f;
  for (int i = threadIdx.x; i < 1024; i += 256) s += partials[i];
  sm[threadIdx.x] = s;
  __syncthreads();
#pragma unroll
  for (int r = 128; r > 0; r >>= 1) {
    if (threadIdx.x < r) sm[threadIdx.x] += sm[threadIdx.x + r];
    __syncthreads();
  }
  if (threadIdx.x == 0) out[0] = sm[0] * (1.0f / (float)(B_ * E_));
}

extern "C" void kernel_launch(void* const* d_in, const int* in_sizes, int n_in,
                              void* d_out, int out_size, void* d_ws, size_t ws_size,
                              hipStream_t stream) {
  (void)in_sizes; (void)n_in; (void)out_size; (void)ws_size;
  const float* x    = (const float*)d_in[0];  // (B,T,E)
  const float* tgt  = (const float*)d_in[1];  // (B,E)
  const float* W_ih = (const float*)d_in[2];  // (L,4E,E)
  const float* W_hh = (const float*)d_in[3];  // (L,4E,E)
  float* out = (float*)d_out;                 // [loss(1), h(B*E), c(B*E)]
  float* ws  = (float*)d_ws;

  const size_t BE = (size_t)B_ * E_;
  float* hb[2] = {ws,          ws + 2 * BE};
  float* cb[2] = {ws + BE,     ws + 3 * BE};
  float* partials = ws + 4 * BE;  // 1024 floats

  lstm_init_kernel<<<(B_ * E_) / 256, 256, 0, stream>>>(x, hb[0], cb[0]);

  const size_t wstep = (size_t)4 * E_ * E_;
  for (int t = 0; t < L_; ++t) {
    const int p = t & 1;
    lstm_step_kernel<<<1024, 256, 0, stream>>>(
        x + (size_t)t * E_,
        W_ih + (size_t)t * wstep,
        W_hh + (size_t)t * wstep,
        hb[p], cb[p], hb[1 - p], cb[1 - p]);
  }

  const int fp = (L_ & 1) ? 1 : 0;  // buffer holding the final state
  lstm_finalize_kernel<<<1024, 256, 0, stream>>>(hb[fp], cb[fp], tgt, out, partials);
  lstm_reduce_kernel<<<1, 256, 0, stream>>>(partials, out);
}
